// JepaWorldModel_85255100826148
// MI455X (gfx1250) — compile-verified, matrix operand order checked by reading, not run on hardware
//
#include <hip/hip_runtime.h>
#include <hip/hip_bf16.h>

typedef __attribute__((ext_vector_type(16))) __bf16 v16bf;
typedef __attribute__((ext_vector_type(8)))  __bf16 v8bf;
typedef __attribute__((ext_vector_type(8)))  float  v8f;

#define AS3 __attribute__((address_space(3)))

static constexpr int B_ = 64, T_ = 512, D_ = 256, FF_ = 512, L_ = 4, H_ = 8;
static constexpr int NTOK = B_ * T_;

__device__ __forceinline__ v8f wmma_bf16(v16bf a, v16bf b, v8f c) {
  // D = A(16x32 bf16) * B(32x16 bf16) + C(16x16 f32)
  return __builtin_amdgcn_wmma_f32_16x16x32_bf16(false, a, false, b, (short)0, c,
                                                 false, false);
}

// Per-lane 16-element bf16 fragment: two contiguous runs of 8
// (k = base..base+7 and k = base+16..base+23); lanes 0-15 use base s=0,
// lanes 16-31 use s=8, matching the ISA 16-bit A/B VGPR layout.
__device__ __forceinline__ v16bf load_frag(const __bf16* p) {
  v8bf lo = *(const v8bf*)(p);
  v8bf hi = *(const v8bf*)(p + 16);
  v16bf r;
#pragma unroll
  for (int i = 0; i < 8; ++i) { r[i] = lo[i]; r[8 + i] = hi[i]; }
  return r;
}

// Async global -> LDS copy of 8 bytes per lane (VGLOBAL encoding,
// ASYNCcnt-tracked). ldsaddr = low 32 bits of the addrspace(3) pointer.
__device__ __forceinline__ void async_copy_b64(unsigned ldsaddr,
                                               const void* gaddr) {
  asm volatile("global_load_async_to_lds_b64 %0, %1, off"
               :
               : "v"(ldsaddr), "v"((unsigned long long)(uintptr_t)gaddr)
               : "memory");
}
__device__ __forceinline__ void wait_async0() {
  asm volatile("s_wait_asynccnt 0x0" ::: "memory");
}

// ---------------------------------------------------------------- embeddings
__global__ void __launch_bounds__(256)
embed_k(const float* __restrict__ lat, const int* __restrict__ mv,
        const float* __restrict__ pos, const float* __restrict__ ef,
        const float* __restrict__ et, const float* __restrict__ ep,
        float* __restrict__ x) {
  const int n = blockIdx.x, d = threadIdx.x;
  const int id = mv[n];
  const int promo = id / 4096, rem = id % 4096, frm = rem / 64, to = rem % 64;
  const int t = n % T_;
  x[(size_t)n * D_ + d] = lat[(size_t)n * D_ + d] + pos[t * D_ + d] +
                          ef[frm * D_ + d] + et[to * D_ + d] + ep[promo * D_ + d];
}

// ------------------------------------------------------------ f32 -> bf16
__global__ void __launch_bounds__(256)
cvt_bf16(const float* __restrict__ s, __bf16* __restrict__ d, int n) {
  int i = blockIdx.x * 256 + threadIdx.x;
  if (i < n) d[i] = (__bf16)s[i];
}

// -------------------------------------------------- LayerNorm -> bf16 rows
__global__ void __launch_bounds__(256)
ln_bf16(const float* __restrict__ x, const float* __restrict__ g,
        const float* __restrict__ bt, __bf16* __restrict__ y) {
  const int lane = threadIdx.x & 31, widx = threadIdx.x >> 5;
  const int row = blockIdx.x * 8 + widx;
  const float* xr = x + (size_t)row * D_ + lane * 8;
  float v[8];
  *(float4*)&v[0] = *(const float4*)xr;
  *(float4*)&v[4] = *(const float4*)(xr + 4);
  float sm = 0.f;
#pragma unroll
  for (int i = 0; i < 8; ++i) sm += v[i];
#pragma unroll
  for (int off = 16; off >= 1; off >>= 1) sm += __shfl_xor(sm, off, 32);
  const float mu = sm * (1.f / D_);
  float sq = 0.f;
#pragma unroll
  for (int i = 0; i < 8; ++i) { float dd = v[i] - mu; sq += dd * dd; }
#pragma unroll
  for (int off = 16; off >= 1; off >>= 1) sq += __shfl_xor(sq, off, 32);
  const float rs = rsqrtf(sq * (1.f / D_) + 1e-5f);
  v8bf ov;
#pragma unroll
  for (int i = 0; i < 8; ++i) {
    int c = lane * 8 + i;
    ov[i] = (__bf16)((v[i] - mu) * rs * g[c] + bt[c]);
  }
  *(v8bf*)(y + (size_t)row * D_ + lane * 8) = ov;
}

// ------------------------------------- final LayerNorm + mask-select, f32 out
__global__ void __launch_bounds__(256)
ln_final(const float* __restrict__ x, const float* __restrict__ g,
         const float* __restrict__ bt, const unsigned char* __restrict__ mask,
         const float* __restrict__ lat, float* __restrict__ out) {
  const int lane = threadIdx.x & 31, widx = threadIdx.x >> 5;
  const int row = blockIdx.x * 8 + widx;
  const float* xr = x + (size_t)row * D_ + lane * 8;
  float v[8];
  *(float4*)&v[0] = *(const float4*)xr;
  *(float4*)&v[4] = *(const float4*)(xr + 4);
  float sm = 0.f;
#pragma unroll
  for (int i = 0; i < 8; ++i) sm += v[i];
#pragma unroll
  for (int off = 16; off >= 1; off >>= 1) sm += __shfl_xor(sm, off, 32);
  const float mu = sm * (1.f / D_);
  float sq = 0.f;
#pragma unroll
  for (int i = 0; i < 8; ++i) { float dd = v[i] - mu; sq += dd * dd; }
#pragma unroll
  for (int off = 16; off >= 1; off >>= 1) sq += __shfl_xor(sq, off, 32);
  const float rs = rsqrtf(sq * (1.f / D_) + 1e-5f);
  const bool keep = mask[row] != 0;
  const size_t base = (size_t)row * D_;
#pragma unroll
  for (int i = 0; i < 8; ++i) {
    int c = lane * 8 + i;
    float lv = (v[i] - mu) * rs * g[c] + bt[c];
    out[base + c] = keep ? lv : lat[base + c];
  }
}

// ----------------------------------------------------------- WMMA GEMM
// C[M,N] = A[M,K](bf16) * W[N,K]^T(bf16) + bias, epilogue by mode:
//   0: store bf16   1: relu -> bf16   2: += residual(f32) -> f32
// Block: 128 thr = 4 waves; block tile 16(M) x 128(N); wave tile 16x32
// (two WMMA accumulators share one A fragment). A tile (16x32 bf16) staged
// to LDS with GLOBAL_LOAD_ASYNC_TO_LDS (ASYNCcnt path).
__global__ void __launch_bounds__(128)
gemm_bf16(const __bf16* __restrict__ A, const __bf16* __restrict__ W,
          const float* __restrict__ bias, int K, int N, int mode,
          const float* __restrict__ resid, float* __restrict__ outf,
          __bf16* __restrict__ outb) {
  __shared__ __bf16 aT[16 * 32];
  const int lane = threadIdx.x & 31;
  const int widx = threadIdx.x >> 5;
  const int l16 = lane & 15, lh = lane >> 4, s = lh * 8;
  const int rowBase = blockIdx.x * 16;
  const int colBase = blockIdx.y * 128 + widx * 32;
  const __bf16* wrow0 = W + (size_t)(colBase + l16) * K;
  const __bf16* wrow1 = W + (size_t)(colBase + 16 + l16) * K;
  const int str = threadIdx.x >> 3;            // staging: row 0..15
  const int stc = (threadIdx.x & 7) * 4;       // staging: 4 bf16 = 8B
  const __bf16* gsrc = A + (size_t)(rowBase + str) * K + stc;
  const unsigned ldsaddr =
      (unsigned)(uintptr_t)(AS3 __bf16*)&aT[str * 32 + stc];
  v8f acc0 = {}, acc1 = {};
  for (int kb = 0; kb < K; kb += 32) {
    __syncthreads();
    // async memory -> LDS copy, no VGPR round trip (tracked by ASYNCcnt)
    async_copy_b64(ldsaddr, gsrc + kb);
    wait_async0();
    __syncthreads();
    if (kb + 32 < K) {
      __builtin_prefetch(wrow0 + kb + 32, 0, 1);  // global_prefetch_b8
      __builtin_prefetch(wrow1 + kb + 32, 0, 1);
    }
    v16bf af = load_frag(&aT[l16 * 32 + s]);
    acc0 = wmma_bf16(af, load_frag(wrow0 + kb + s), acc0);
    acc1 = wmma_bf16(af, load_frag(wrow1 + kb + s), acc1);
  }
  const int col0 = colBase + l16;
  const float bv0 = bias[col0];
  const float bv1 = bias[col0 + 16];
#pragma unroll
  for (int r = 0; r < 8; ++r) {
    const int grow = rowBase + r + s;  // C layout: M = r + 8*(lane>=16)
    const size_t idx0 = (size_t)grow * N + col0;
    const size_t idx1 = idx0 + 16;
    float v0 = acc0[r] + bv0;
    float v1 = acc1[r] + bv1;
    if (mode == 0) {
      outb[idx0] = (__bf16)v0;
      outb[idx1] = (__bf16)v1;
    } else if (mode == 1) {
      outb[idx0] = (__bf16)fmaxf(v0, 0.f);
      outb[idx1] = (__bf16)fmaxf(v1, 0.f);
    } else {
      outf[idx0] = resid[idx0] + v0;
      outf[idx1] = resid[idx1] + v1;
    }
  }
}

// ------------------------------------------------ flash attention (WMMA)
// One wave per (batch, head, 16-query tile). Block-causal mask at 8-token
// granularity => only the diagonal 16x16 tile needs masking (rows 0-7 cannot
// see cols 8-15). qkv layout: [token][768] = Q|K|V, head h at cols h*32.
__global__ void __launch_bounds__(128)
attn_k(const __bf16* __restrict__ qkv, __bf16* __restrict__ out) {
  const int lane = threadIdx.x & 31;
  const int widx = threadIdx.x >> 5;
  const int l16 = lane & 15, lh = lane >> 4, s = lh * 8;
  const int wid = blockIdx.x * 4 + widx;
  const int qt = wid & 31;
  const int h = (wid >> 5) & 7;
  const int b = wid >> 8;

  __shared__ __bf16 plds_all[4][16 * 32];
  __bf16* plds = plds_all[widx];

  const size_t tokBase = (size_t)b * T_;
  const __bf16* qrow = qkv + (tokBase + qt * 16 + l16) * 768 + h * 32;
  const v16bf aq = load_frag(qrow + s);

  v8f o0 = {}, o1 = {};
  float mrow[8], lrow[8];
#pragma unroll
  for (int r = 0; r < 8; ++r) { mrow[r] = -3.0e38f; lrow[r] = 0.f; }

  const int ktiles = qt + 1;           // 16-key tiles visible to this q-tile
  const int chunks = (ktiles + 1) >> 1;
  const float scl = 0.17677669529663687f;  // 1/sqrt(32)

  for (int c = 0; c < chunks; ++c) {
    const int k0 = c * 32;
    const bool have1 = (2 * c + 1) < ktiles;
    v8f s0, s1 = {};
    {
      const __bf16* kb = qkv + (tokBase + k0 + l16) * 768 + 256 + h * 32;
      v8f z = {};
      s0 = wmma_bf16(aq, load_frag(kb + s), z);
    }
    if (have1) {
      const __bf16* kb = qkv + (tokBase + k0 + 16 + l16) * 768 + 256 + h * 32;
      v8f z = {};
      s1 = wmma_bf16(aq, load_frag(kb + s), z);
    }
    const bool mask0 = (2 * c == ktiles - 1) && (lh == 0) && (l16 >= 8);
    const bool mask1 = (2 * c + 1 == ktiles - 1) && (lh == 0) && (l16 >= 8);
#pragma unroll
    for (int r = 0; r < 8; ++r) {
      float a0 = mask0 ? -3.0e38f : s0[r] * scl;
      float a1 = (!have1 || mask1) ? -3.0e38f : s1[r] * scl;
      float mx = fmaxf(a0, a1);
#pragma unroll
      for (int off = 8; off >= 1; off >>= 1)
        mx = fmaxf(mx, __shfl_xor(mx, off, 16));
      const float mnew = fmaxf(mrow[r], mx);
      const float alpha = __expf(mrow[r] - mnew);
      const float p0 = __expf(a0 - mnew);
      const float p1 = __expf(a1 - mnew);
      float rs = p0 + p1;
#pragma unroll
      for (int off = 8; off >= 1; off >>= 1) rs += __shfl_xor(rs, off, 16);
      lrow[r] = lrow[r] * alpha + rs;
      mrow[r] = mnew;
      o0[r] *= alpha;
      o1[r] *= alpha;
      plds[(r + s) * 32 + l16] = (__bf16)p0;        // C-layout -> LDS
      plds[(r + s) * 32 + 16 + l16] = (__bf16)p1;
    }
    // intra-wave LDS RAW: DS ops are in-order per wave; wait for visibility
    asm volatile("s_wait_dscnt 0x0" ::: "memory");
    const v16bf ap = load_frag(&plds[l16 * 32 + s]);  // A-layout P tile
    const __bf16* vb = qkv + (tokBase + k0) * 768 + 512 + h * 32 + l16;
    v16bf bv0, bv1;
#pragma unroll
    for (int i = 0; i < 8; ++i) {
      bv0[i] = vb[(size_t)(s + i) * 768];
      bv0[8 + i] = vb[(size_t)(16 + s + i) * 768];
      bv1[i] = vb[(size_t)(s + i) * 768 + 16];
      bv1[8 + i] = vb[(size_t)(16 + s + i) * 768 + 16];
    }
    o0 = wmma_bf16(ap, bv0, o0);
    o1 = wmma_bf16(ap, bv1, o1);
  }
  const size_t orow0 = (tokBase + qt * 16) * D_ + h * 32;
#pragma unroll
  for (int r = 0; r < 8; ++r) {
    const float inv = 1.f / lrow[r];
    const size_t base = orow0 + (size_t)(r + s) * D_;
    out[base + l16] = (__bf16)(o0[r] * inv);
    out[base + 16 + l16] = (__bf16)(o1[r] * inv);
  }
}

// ------------------------------------------------------------------ host
extern "C" void kernel_launch(void* const* d_in, const int* in_sizes, int n_in,
                              void* d_out, int out_size, void* d_ws,
                              size_t ws_size, hipStream_t stream) {
  (void)in_sizes; (void)n_in; (void)out_size; (void)ws_size;
  const float* latents = (const float*)d_in[0];
  const int* move_ids = (const int*)d_in[1];
  const unsigned char* mask = (const unsigned char*)d_in[2];
  const float* positional = (const float*)d_in[3];
  const float* ef = (const float*)d_in[4];
  const float* et = (const float*)d_in[5];
  const float* ep = (const float*)d_in[6];
  const float* wqkv = (const float*)d_in[7];
  const float* bqkv = (const float*)d_in[8];
  const float* wo = (const float*)d_in[9];
  const float* bo = (const float*)d_in[10];
  const float* w1 = (const float*)d_in[11];
  const float* b1 = (const float*)d_in[12];
  const float* w2 = (const float*)d_in[13];
  const float* b2 = (const float*)d_in[14];
  const float* ng = (const float*)d_in[15];
  const float* nb = (const float*)d_in[16];
  float* out = (float*)d_out;

  char* p = (char*)d_ws;
  auto alloc = [&](size_t bytes) {
    char* r = p;
    p += (bytes + 255) & ~(size_t)255;
    return r;
  };
  float* xf = (float*)alloc((size_t)NTOK * D_ * 4);
  __bf16* y = (__bf16*)alloc((size_t)NTOK * D_ * 2);
  __bf16* qkv = (__bf16*)alloc((size_t)NTOK * (3 * D_) * 2);
  __bf16* at = (__bf16*)alloc((size_t)NTOK * D_ * 2);
  __bf16* h1 = (__bf16*)alloc((size_t)NTOK * FF_ * 2);
  __bf16* wqb = (__bf16*)alloc((size_t)L_ * 3 * D_ * D_ * 2);
  __bf16* wob = (__bf16*)alloc((size_t)L_ * D_ * D_ * 2);
  __bf16* w1b = (__bf16*)alloc((size_t)L_ * FF_ * D_ * 2);
  __bf16* w2b = (__bf16*)alloc((size_t)L_ * D_ * FF_ * 2);

  const int nwq = L_ * 3 * D_ * D_, nwo = L_ * D_ * D_;
  const int nw1 = L_ * FF_ * D_, nw2 = L_ * D_ * FF_;
  cvt_bf16<<<(nwq + 255) / 256, 256, 0, stream>>>(wqkv, wqb, nwq);
  cvt_bf16<<<(nwo + 255) / 256, 256, 0, stream>>>(wo, wob, nwo);
  cvt_bf16<<<(nw1 + 255) / 256, 256, 0, stream>>>(w1, w1b, nw1);
  cvt_bf16<<<(nw2 + 255) / 256, 256, 0, stream>>>(w2, w2b, nw2);

  embed_k<<<NTOK, 256, 0, stream>>>(latents, move_ids, positional, ef, et, ep, xf);

  for (int l = 0; l < L_; ++l) {
    ln_bf16<<<NTOK / 8, 256, 0, stream>>>(xf, ng, nb, y);
    gemm_bf16<<<dim3(NTOK / 16, (3 * D_) / 128), 128, 0, stream>>>(
        y, wqb + (size_t)l * 3 * D_ * D_, bqkv + l * 3 * D_, D_, 3 * D_, 0,
        nullptr, nullptr, qkv);
    attn_k<<<(B_ * H_ * (T_ / 16)) / 4, 128, 0, stream>>>(qkv, at);
    gemm_bf16<<<dim3(NTOK / 16, D_ / 128), 128, 0, stream>>>(
        at, wob + (size_t)l * D_ * D_, bo + l * D_, D_, D_, 2, xf, xf, nullptr);
    ln_bf16<<<NTOK / 8, 256, 0, stream>>>(xf, ng, nb, y);
    gemm_bf16<<<dim3(NTOK / 16, FF_ / 128), 128, 0, stream>>>(
        y, w1b + (size_t)l * FF_ * D_, b1 + l * FF_, D_, FF_, 1, nullptr,
        nullptr, h1);
    gemm_bf16<<<dim3(NTOK / 16, D_ / 128), 128, 0, stream>>>(
        h1, w2b + (size_t)l * D_ * FF_, b2 + l * D_, FF_, D_, 2, xf, xf,
        nullptr);
  }
  ln_final<<<NTOK / 8, 256, 0, stream>>>(xf, ng, nb, mask, latents, out);
}
